// TimeConv_71124658422185
// MI455X (gfx1250) — matrix operand order repaired and checked by reading, not compiled, not run on hardware
//
#include <hip/hip_runtime.h>
#include <hip/hip_bf16.h>

typedef __attribute__((ext_vector_type(16))) _Float16 v16h;
typedef __attribute__((ext_vector_type(8)))  _Float16 v8h;
typedef __attribute__((ext_vector_type(8)))  float    v8f;

#define NEG_SLOPE 0.1f

// ---------------------------------------------------------------------------
// Fragment loaders using this file's contiguous k order (k = 16h + s) for both A and B:
// A (16x32, f16): lane L (m=L%16, h=L/16) holds K = {c*32+h*16 .. +7} in V0..V3
//                 and K = {c*32+h*16+8 .. +7} in V4..V7  -> two 16B segments.
// B (32x16, f16): lane L (n=L%16, h=L/16) holds K = c*32+h*16 .. +15
//                 -> one contiguous 32B segment when weights are stored [N][K].
// C/D (16x16,f32): lane L (n=L%16, h=L/16), VGPR r holds row m = r + 8*h.
// ---------------------------------------------------------------------------
__device__ inline v16h load_a_frag(const _Float16* xrow_base, int kpad,
                                   int m, int h, int kbase) {
  const _Float16* p = xrow_base + (size_t)m * kpad + kbase + h * 16;
  v8h lo = *(const v8h*)p;
  v8h hi = *(const v8h*)(p + 8);
  v16h a;
#pragma unroll
  for (int i = 0; i < 8; ++i) { a[i] = lo[i]; a[8 + i] = hi[i]; }
  return a;
}

__device__ inline v16h load_b_frag(const _Float16* wt, int kpad,
                                   int n, int h, int kbase) {
  const _Float16* p = wt + (size_t)n * kpad + kbase + h * 16;
  return *(const v16h*)p;
}

// ---------------------------------------------------------------------------
// Generic node MLP: y = LeakyReLU(x @ W1 + b1) @ W2 + b2
//   x   : [Npad][kpad]  f16 (kpad multiple of 32)
//   w1t : [64][kpad]    f16 (transposed, zero-padded)
//   w2t : [128][64]     f16 (transposed)
// Output: f32 [node][128] (out16==null) or f16 at out16[node*stride+off+n].
// 256 threads = 8 waves; each wave owns a 16-node tile -> 128 nodes/block.
// ---------------------------------------------------------------------------
__global__ __launch_bounds__(256) void mlp_wmma_kernel(
    const _Float16* __restrict__ x, int kpad,
    const _Float16* __restrict__ w1t, const float* __restrict__ b1,
    const _Float16* __restrict__ w2t, const float* __restrict__ b2,
    float* __restrict__ out32, _Float16* __restrict__ out16,
    int out_stride, int out_off) {
  __shared__ _Float16 s_hid[8][16 * 64];
  const int wid  = threadIdx.x >> 5;
  const int lane = threadIdx.x & 31;
  const int lm   = lane & 15;
  const int h    = lane >> 4;
  const int base = blockIdx.x * 128 + wid * 16;   // first node of this tile

  // ---- stage 1: [16 x kpad] @ [kpad x 64] ----
  v8f acc[4] = {};
  const _Float16* xt = x + (size_t)base * kpad;
  const int kch = kpad >> 5;
  for (int c = 0; c < kch; ++c) {
    v16h a = load_a_frag(xt, kpad, lm, h, c * 32);
#pragma unroll
    for (int t = 0; t < 4; ++t) {
      v16h b = load_b_frag(w1t, kpad, t * 16 + lm, h, c * 32);
      acc[t] = __builtin_amdgcn_wmma_f32_16x16x32_f16(
          false, a, false, b, (short)0, acc[t], false, false);
    }
  }
  // bias + leaky-relu, stage hidden activations as f16 in LDS
#pragma unroll
  for (int t = 0; t < 4; ++t) {
    const int n = t * 16 + lm;
    const float bb = b1[n];
#pragma unroll
    for (int r = 0; r < 8; ++r) {
      float y = acc[t][r] + bb;
      y = (y > 0.0f) ? y : NEG_SLOPE * y;
      s_hid[wid][(r + 8 * h) * 64 + n] = (_Float16)y;
    }
  }
  __syncthreads();

  // ---- stage 2: [16 x 64] @ [64 x 128] ----
  v8f acc2[8] = {};
#pragma unroll
  for (int c = 0; c < 2; ++c) {
    v16h a = load_a_frag(&s_hid[wid][0], 64, lm, h, c * 32);
#pragma unroll
    for (int t = 0; t < 8; ++t) {
      v16h b = load_b_frag(w2t, 64, t * 16 + lm, h, c * 32);
      acc2[t] = __builtin_amdgcn_wmma_f32_16x16x32_f16(
          false, a, false, b, (short)0, acc2[t], false, false);
    }
  }
#pragma unroll
  for (int t = 0; t < 8; ++t) {
    const int n = t * 16 + lm;
    const float bb = b2[n];
#pragma unroll
    for (int r = 0; r < 8; ++r) {
      const int node = base + r + 8 * h;            // rows padded to Npad
      const float y = acc2[t][r] + bb;
      if (out32) out32[(size_t)node * 128 + n] = y;
      else       out16[(size_t)node * out_stride + out_off + n] = (_Float16)y;
    }
  }
}

// ---------------------------------------------------------------------------
// Final MLP: hidden = LeakyReLU(x_o @ o_w1 + o_b1) [16x128], out = hidden @ o_w2 + o_b2
// ---------------------------------------------------------------------------
__global__ __launch_bounds__(256) void out_wmma_kernel(
    const _Float16* __restrict__ xo,   // [Npad][256] f16
    const _Float16* __restrict__ w1t,  // [128][256]  f16
    const float* __restrict__ b1,      // [128]
    const float* __restrict__ w2,      // [128]
    const float* __restrict__ b2,      // [1]
    float* __restrict__ out, int N) {
  __shared__ float s_red[8][32][8];
  const int wid  = threadIdx.x >> 5;
  const int lane = threadIdx.x & 31;
  const int lm   = lane & 15;
  const int h    = lane >> 4;
  const int base = blockIdx.x * 128 + wid * 16;

  v8f acc[8] = {};
  const _Float16* xt = xo + (size_t)base * 256;
#pragma unroll
  for (int c = 0; c < 8; ++c) {
    v16h a = load_a_frag(xt, 256, lm, h, c * 32);
#pragma unroll
    for (int t = 0; t < 8; ++t) {
      v16h b = load_b_frag(w1t, 256, t * 16 + lm, h, c * 32);
      acc[t] = __builtin_amdgcn_wmma_f32_16x16x32_f16(
          false, a, false, b, (short)0, acc[t], false, false);
    }
  }
  // bias + leaky + dot with o_w2, partial per lane over its 8 rows
  float s[8] = {};
#pragma unroll
  for (int t = 0; t < 8; ++t) {
    const int n = t * 16 + lm;
    const float bb = b1[n];
    const float ww = w2[n];
#pragma unroll
    for (int r = 0; r < 8; ++r) {
      float y = acc[t][r] + bb;
      y = (y > 0.0f) ? y : NEG_SLOPE * y;
      s[r] += y * ww;
    }
  }
#pragma unroll
  for (int r = 0; r < 8; ++r) s_red[wid][lane][r] = s[r];
  __syncthreads();
  if (lane < 16) {
    const int m = lane, hh = m >> 3, r = m & 7;
    float sum = 0.0f;
#pragma unroll
    for (int i = 0; i < 16; ++i) sum += s_red[wid][hh * 16 + i][r];
    const int node = base + m;
    if (node < N) out[node] = sum + b2[0];
  }
}

// ---------------------------------------------------------------------------
// Edge scatter: one wave per edge; 4 floats per lane; f32 atomics into L2.
// ---------------------------------------------------------------------------
__global__ __launch_bounds__(256) void scatter_kernel(
    const float* __restrict__ h0, const int* __restrict__ src,
    const int* __restrict__ dst, const float* __restrict__ pos,
    float* __restrict__ sum, float* __restrict__ cnt,
    float* __restrict__ pos_sum, int E) {
  const int wid  = threadIdx.x >> 5;
  const int lane = threadIdx.x & 31;
  const int e = blockIdx.x * 8 + wid;
  if (e >= E) return;
  const int s = src[e], d = dst[e];
  const float4 v = *(const float4*)(h0 + (size_t)s * 128 + lane * 4);
  float* dp = sum + (size_t)d * 128 + lane * 4;
  atomicAdd(dp + 0, v.x); atomicAdd(dp + 1, v.y);
  atomicAdd(dp + 2, v.z); atomicAdd(dp + 3, v.w);
  if (lane == 0) {
    atomicAdd(cnt + d, 1.0f);
    if (pos) atomicAdd(pos_sum + d, pos[e]);
  }
}

// ---------------------------------------------------------------------------
// Elementwise build / convert kernels
// ---------------------------------------------------------------------------
__global__ void wconv_kernel(const float* __restrict__ src, _Float16* __restrict__ dst,
                             int K, int Kpad, int Nout) {
  int idx = blockIdx.x * blockDim.x + threadIdx.x;
  if (idx >= Nout * Kpad) return;
  const int n = idx / Kpad, k = idx % Kpad;
  dst[idx] = (k < K) ? (_Float16)src[(size_t)k * Nout + n] : (_Float16)0.0f;
}

__global__ void build_pad32_kernel(const float* __restrict__ src, int width,
                                   _Float16* __restrict__ dst, int N) {
  int idx = blockIdx.x * blockDim.x + threadIdx.x;
  if (idx >= N * 32) return;
  const int node = idx >> 5, c = idx & 31;
  dst[idx] = (c < width) ? (_Float16)src[(size_t)node * width + c] : (_Float16)0.0f;
}

__global__ void build_xnm_kernel(const float* __restrict__ sum_m,
                                 const float* __restrict__ cnt_m,
                                 const float* __restrict__ pos_sum,
                                 const float* __restrict__ feat,
                                 _Float16* __restrict__ x, int N) {
  int idx = blockIdx.x * blockDim.x + threadIdx.x;
  if (idx >= N * 224) return;
  const int node = idx / 224, c = idx % 224;
  float v;
  if (c < 128) {
    v = sum_m[(size_t)node * 128 + c] / fmaxf(cnt_m[node], 1.0f);
  } else if (c == 128) {
    v = pos_sum[node] / fmaxf(cnt_m[node], 1.0f);
  } else if (c < 193) {
    v = feat[(size_t)node * 64 + (c - 129)];
  } else {
    v = 0.0f;
  }
  x[idx] = (_Float16)v;
}

__global__ void build_xng_kernel(const float* __restrict__ sum_g,
                                 const float* __restrict__ cnt_g,
                                 const float* __restrict__ feat,
                                 _Float16* __restrict__ x, int N) {
  int idx = blockIdx.x * blockDim.x + threadIdx.x;
  if (idx >= N * 192) return;
  const int node = idx / 192, c = idx % 192;
  float v = (c < 128) ? sum_g[(size_t)node * 128 + c] / fmaxf(cnt_g[node], 1.0f)
                      : feat[(size_t)node * 64 + (c - 128)];
  x[idx] = (_Float16)v;
}

__global__ void select_kernel(const float* __restrict__ h_m,
                              const float* __restrict__ h_g,
                              const int* __restrict__ is_po,
                              const int* __restrict__ is_module,
                              _Float16* __restrict__ x_o, int N) {
  int idx = blockIdx.x * blockDim.x + threadIdx.x;
  if (idx >= N * 128) return;
  const int node = idx >> 7, c = idx & 127;
  float m = h_m[idx], g = h_g[idx];
  if (is_po[node] != 1) { m = fmaxf(m, 0.0f); g = fmaxf(g, 0.0f); }
  const float hh = (is_module[node] == 1) ? m : g;
  x_o[(size_t)node * 256 + c] = (_Float16)hh;
}

// ---------------------------------------------------------------------------
extern "C" void kernel_launch(void* const* d_in, const int* in_sizes, int n_in,
                              void* d_out, int out_size, void* d_ws, size_t ws_size,
                              hipStream_t stream) {
  const float* feat   = (const float*)d_in[0];
  const float* pi_ft  = (const float*)d_in[1];
  const float* level  = (const float*)d_in[2];
  const float* bitpos = (const float*)d_in[3];
  const int*   is_po  = (const int*)d_in[4];
  const int*   is_mod = (const int*)d_in[5];
  const int*   src_m  = (const int*)d_in[6];
  const int*   dst_m  = (const int*)d_in[7];
  const int*   src_g  = (const int*)d_in[8];
  const int*   dst_g  = (const int*)d_in[9];
  const float* pi_w1 = (const float*)d_in[10]; const float* pi_b1 = (const float*)d_in[11];
  const float* pi_w2 = (const float*)d_in[12]; const float* pi_b2 = (const float*)d_in[13];
  const float* nm_w1 = (const float*)d_in[14]; const float* nm_b1 = (const float*)d_in[15];
  const float* nm_w2 = (const float*)d_in[16]; const float* nm_b2 = (const float*)d_in[17];
  const float* ng_w1 = (const float*)d_in[18]; const float* ng_b1 = (const float*)d_in[19];
  const float* ng_w2 = (const float*)d_in[20]; const float* ng_b2 = (const float*)d_in[21];
  const float* g_w1  = (const float*)d_in[22]; const float* g_b1  = (const float*)d_in[23];
  const float* g_w2  = (const float*)d_in[24]; const float* g_b2  = (const float*)d_in[25];
  const float* o_w1  = (const float*)d_in[26]; const float* o_b1  = (const float*)d_in[27];
  const float* o_w2  = (const float*)d_in[28]; const float* o_b2  = (const float*)d_in[29];
  float* out = (float*)d_out;

  const int N = in_sizes[4];          // is_po count
  const int E = in_sizes[6];          // src_m count
  const size_t Npad = (size_t)((N + 127) / 128) * 128;

  // ---- workspace carve-out (256B aligned slabs) ----
  char* ws = (char*)d_ws;
  size_t off = 0;
  auto take = [&](size_t elems, size_t esz) -> void* {
    void* p = ws + off;
    off += ((elems * esz + 255) & ~(size_t)255);
    return p;
  };
  float* h0      = (float*)take(Npad * 128, 4);
  float* sum_m   = (float*)take(Npad * 128, 4);
  float* sum_g   = (float*)take(Npad * 128, 4);
  float* cnt_m   = (float*)take(Npad, 4);
  float* cnt_g   = (float*)take(Npad, 4);
  float* pos_sum = (float*)take(Npad, 4);
  float* h_m     = (float*)take(Npad * 128, 4);
  float* h_g     = (float*)take(Npad * 128, 4);
  _Float16* x_pi = (_Float16*)take(Npad * 32, 2);
  _Float16* x_nm = (_Float16*)take(Npad * 224, 2);
  _Float16* x_ng = (_Float16*)take(Npad * 192, 2);
  _Float16* x_g  = (_Float16*)take(Npad * 32, 2);
  _Float16* x_o  = (_Float16*)take(Npad * 256, 2);
  _Float16* w1t_pi = (_Float16*)take(64 * 32, 2);
  _Float16* w2t_pi = (_Float16*)take(128 * 64, 2);
  _Float16* w1t_nm = (_Float16*)take(64 * 224, 2);
  _Float16* w2t_nm = (_Float16*)take(128 * 64, 2);
  _Float16* w1t_ng = (_Float16*)take(64 * 192, 2);
  _Float16* w2t_ng = (_Float16*)take(128 * 64, 2);
  _Float16* w1t_g  = (_Float16*)take(64 * 32, 2);
  _Float16* w2t_g  = (_Float16*)take(128 * 64, 2);
  _Float16* w1t_o  = (_Float16*)take(128 * 256, 2);

  // ---- zero accumulation buffers (graph-capturable) ----
  hipMemsetAsync(sum_m, 0, Npad * 128 * sizeof(float), stream);
  hipMemsetAsync(sum_g, 0, Npad * 128 * sizeof(float), stream);
  hipMemsetAsync(cnt_m, 0, Npad * sizeof(float), stream);
  hipMemsetAsync(cnt_g, 0, Npad * sizeof(float), stream);
  hipMemsetAsync(pos_sum, 0, Npad * sizeof(float), stream);

  auto cdiv = [](long long a, long long b) { return (int)((a + b - 1) / b); };
  auto wconv = [&](const float* src, _Float16* dst, int K, int Kpad, int Nout) {
    wconv_kernel<<<cdiv((long long)Nout * Kpad, 256), 256, 0, stream>>>(src, dst, K, Kpad, Nout);
  };

  // ---- weight convert + transpose into padded f16 [Nout][Kpad] ----
  wconv(pi_w1, w1t_pi,   4,  32,  64);
  wconv(pi_w2, w2t_pi,  64,  64, 128);
  wconv(nm_w1, w1t_nm, 193, 224,  64);
  wconv(nm_w2, w2t_nm,  64,  64, 128);
  wconv(ng_w1, w1t_ng, 192, 192,  64);
  wconv(ng_w2, w2t_ng,  64,  64, 128);
  wconv(g_w1,  w1t_g,    1,  32,  64);
  wconv(g_w2,  w2t_g,   64,  64, 128);
  wconv(o_w1,  w1t_o,  256, 256, 128);

  const int node_blocks = (int)(Npad / 128);

  // ---- h0 = mlp_pi(pi_feat) ----
  build_pad32_kernel<<<cdiv((long long)N * 32, 256), 256, 0, stream>>>(pi_ft, 4, x_pi, N);
  mlp_wmma_kernel<<<node_blocks, 256, 0, stream>>>(
      x_pi, 32, w1t_pi, pi_b1, w2t_pi, pi_b2, h0, nullptr, 0, 0);

  // ---- edge message passing (segment sums via f32 atomics) ----
  scatter_kernel<<<cdiv(E, 8), 256, 0, stream>>>(h0, src_m, dst_m, bitpos,
                                                 sum_m, cnt_m, pos_sum, E);
  scatter_kernel<<<cdiv(E, 8), 256, 0, stream>>>(h0, src_g, dst_g, nullptr,
                                                 sum_g, cnt_g, nullptr, E);

  // ---- h_m = mlp_nm(cat(neigh_m, pos_m, feat)) ----
  build_xnm_kernel<<<cdiv((long long)N * 224, 256), 256, 0, stream>>>(
      sum_m, cnt_m, pos_sum, feat, x_nm, N);
  mlp_wmma_kernel<<<node_blocks, 256, 0, stream>>>(
      x_nm, 224, w1t_nm, nm_b1, w2t_nm, nm_b2, h_m, nullptr, 0, 0);

  // ---- h_g = mlp_ng(cat(neigh_g, feat)) ----
  build_xng_kernel<<<cdiv((long long)N * 192, 256), 256, 0, stream>>>(
      sum_g, cnt_g, feat, x_ng, N);
  mlp_wmma_kernel<<<node_blocks, 256, 0, stream>>>(
      x_ng, 192, w1t_ng, ng_b1, w2t_ng, ng_b2, h_g, nullptr, 0, 0);

  // ---- hg = mlp_g(level), written f16 directly into x_o[:,128:256] ----
  build_pad32_kernel<<<cdiv((long long)N * 32, 256), 256, 0, stream>>>(level, 1, x_g, N);
  mlp_wmma_kernel<<<node_blocks, 256, 0, stream>>>(
      x_g, 32, w1t_g, g_b1, w2t_g, g_b2, nullptr, x_o, 256, 128);

  // ---- relu-on-non-PO + hetero select -> x_o[:,0:128] ----
  select_kernel<<<cdiv((long long)N * 128, 256), 256, 0, stream>>>(
      h_m, h_g, is_po, is_mod, x_o, N);

  // ---- out = mlp_o(cat(h, hg)) ----
  out_wmma_kernel<<<node_blocks, 256, 0, stream>>>(
      x_o, w1t_o, o_b1, o_w2, o_b2, out, N);
  (void)n_in; (void)out_size; (void)ws_size;
}
